// Dsqn_56453050138933
// MI455X (gfx1250) — compile-verified, matrix operand order checked by reading, not run on hardware
//
#include <hip/hip_runtime.h>

typedef __bf16 bf16_t;
typedef __attribute__((ext_vector_type(16))) __bf16 v16bf;
typedef __attribute__((ext_vector_type(8)))  __bf16 v8bf;
typedef __attribute__((ext_vector_type(4)))  __bf16 v4bf;
typedef __attribute__((ext_vector_type(8)))  float  v8f;
typedef __attribute__((ext_vector_type(4)))  float  v4f;
typedef __attribute__((ext_vector_type(4)))  unsigned int u32x4;
typedef __attribute__((ext_vector_type(8)))  int i32x8;
typedef __attribute__((ext_vector_type(4)))  int i32x4;

#if defined(__has_builtin)
#if __has_builtin(__builtin_amdgcn_tensor_load_to_lds) && __has_builtin(__builtin_amdgcn_s_wait_tensorcnt)
#define HAVE_TDM 1
#endif
#endif
#ifndef HAVE_TDM
#define HAVE_TDM 0
#endif

// ---------- helpers ----------
__device__ __forceinline__ bf16_t f2bf(float f) {
  unsigned u = __builtin_bit_cast(unsigned, f);
  unsigned r = u + 0x7FFFu + ((u >> 16) & 1u);   // round-to-nearest-even
  return __builtin_bit_cast(bf16_t, (unsigned short)(r >> 16));
}
__device__ __forceinline__ bf16_t spike_bf(float v) {      // H(v - 1) as bf16 0/1
  unsigned short us = (v >= 1.0f) ? (unsigned short)0x3F80 : (unsigned short)0;
  return __builtin_bit_cast(bf16_t, us);
}

#if HAVE_TDM
// Tensor DMA: [rows=64, cols=32] bf16 tile of W -> LDS, 8 DWORDs pad every
// 16 DWORDs so each 64B row lands on a 96B (48 bf16) LDS stride.
__device__ __forceinline__ void tdm_load_b_tile(bf16_t* ldsdst, const bf16_t* gsrc,
                                                int K, int N) {
  unsigned ldsb = (unsigned)(unsigned long long)(void*)ldsdst;
  unsigned long long ga = (unsigned long long)(const void*)gsrc;
  u32x4 g0 = { 1u,                                    // count=1 (valid D#)
               ldsb,                                  // lds_addr
               (unsigned)ga,                          // global_addr[31:0]
               (unsigned)((ga >> 32) & 0x01FFFFFFull) | (2u << 30) }; // ga[56:32]|type=2
  i32x8 g1 = { (int)((1u << 16) | (1u << 20) | (3u << 22) | (7u << 25)),
               // data_size=2B | pad_enable | pad_interval=16DW | pad_amount=8DW
               (int)(((unsigned)K & 0xFFFFu) << 16),                     // dim0 lo
               (int)((((unsigned)K >> 16) & 0xFFFFu) |
                     (((unsigned)N & 0xFFFFu) << 16)),                   // dim0 hi | dim1 lo
               (int)((((unsigned)N >> 16) & 0xFFFFu) | (32u << 16)),     // dim1 hi | tile0=32
               64,                                                       // tile_dim1=64
               (int)(unsigned)K,                                         // dim0_stride lo
               0, 0 };
  i32x4 gz = {0, 0, 0, 0};
#if defined(__clang_major__) && (__clang_major__ >= 23)
  i32x8 gz8 = {0, 0, 0, 0, 0, 0, 0, 0};
  __builtin_amdgcn_tensor_load_to_lds(g0, g1, gz, gz, gz8, 0);
#else
  __builtin_amdgcn_tensor_load_to_lds(g0, g1, gz, gz, 0);
#endif
}
#endif

// ---------- fp32 -> bf16 convert, x4 vectorized (n % 4 == 0) ----------
__global__ __launch_bounds__(256) void cvt4_bf16(const float* __restrict__ src,
                                                 bf16_t* __restrict__ dst, int n) {
  int i = (blockIdx.x * 256 + threadIdx.x) * 4;
  if (i >= n) return;
  v4f v = *(const v4f*)(src + i);
  v4bf o;
  #pragma unroll
  for (int e = 0; e < 4; ++e) o[e] = f2bf(v[e]);
  *(v4bf*)(dst + i) = o;
}

// ---------- OIHW fp32 -> [COUT][P*CIN] bf16 (k = p*CIN + c, NHWC k-order) ----------
__global__ __launch_bounds__(256) void permute_cvt(const float* __restrict__ src,
                                                   bf16_t* __restrict__ dst,
                                                   int COUT, int CIN, int P) {
  int idx = blockIdx.x * 256 + threadIdx.x;
  int K = CIN * P;
  if (idx >= COUT * K) return;
  int n = idx / K, r = idx - n * K;
  int p = r / CIN, c = r - p * CIN;
  dst[idx] = f2bf(src[((size_t)n * CIN + c) * P + p]);
}

// ---------- conv1: NCHW bf16 input, 8x8 s4, out [M=204800, 32] (NHWC) ----------
__global__ __launch_bounds__(256) void conv1_wmma(
    const bf16_t* __restrict__ X,  // [512,4,84,84] bf16 copy of x
    const bf16_t* __restrict__ W,  // [32][256], k = c*64 + kh*8 + kw
    const float* __restrict__ bias,
    float* __restrict__ Y)
{
  constexpr int OHW = 400;
  __shared__ alignas(32) bf16_t As[2][128][48];
  __shared__ alignas(32) bf16_t Bs[2][32][48];

  const int tid  = threadIdx.x;
  const int lane = tid & 31, wave = tid >> 5;
  const int l    = lane & 15, half = lane >> 4;
  const int wm   = wave & 3,  wn   = wave >> 2;
  const int m_blk = blockIdx.x * 128;

  // hoisted per-thread u32 element offsets (4 x v4bf A-gathers per step)
  int tmv[4], tkv[4]; unsigned aoff[4];
  #pragma unroll
  for (int j = 0; j < 4; ++j) {
    int idx = tid + j * 256;
    int tm  = idx >> 3;
    int tk0 = (idx & 7) * 4;
    int m   = m_blk + tm;
    int nb  = m / OHW, rp = m - nb * OHW;
    int oh  = rp / 20, ow = rp - oh * 20;
    int po  = (tk0 >> 3) * 84 + (tk0 & 7);
    tmv[j] = tm; tkv[j] = tk0;
    aoff[j] = (unsigned)(nb * 28224 + oh * 4 * 84 + ow * 4 + po);
  }
  const unsigned boff = (unsigned)((tid >> 3) * 256 + (tid & 7) * 4);  // all 256 threads
  const int bsn = tid >> 3, bsk = (tid & 7) * 4;

  v8f zero = {};
  v8f acc0 = zero, acc1 = zero;

  // prologue: stage buffer 0 (ks = 0; soff = 0)
  {
    v4bf ra[4];
    #pragma unroll
    for (int j = 0; j < 4; ++j) ra[j] = *(const v4bf*)(X + aoff[j]);
    v4bf rb = *(const v4bf*)(W + boff);
    #pragma unroll
    for (int j = 0; j < 4; ++j) *(v4bf*)&As[0][tmv[j]][tkv[j]] = ra[j];
    *(v4bf*)&Bs[0][bsn][bsk] = rb;
  }

  for (int ks = 0; ks < 8; ++ks) {
    const int cur = ks & 1, nxt = cur ^ 1;
    __syncthreads();
    v4bf ra[4] = {}; v4bf rb = {};
    const bool more = (ks + 1) < 8;
    if (more) {
      const int k1 = ks + 1;
      const bf16_t* Xp = X + (k1 >> 1) * 7056 + (k1 & 1) * 336;  // uniform scalar base
      const bf16_t* Wp = W + k1 * 32;
      #pragma unroll
      for (int j = 0; j < 4; ++j) ra[j] = *(const v4bf*)(Xp + aoff[j]);
      rb = *(const v4bf*)(Wp + boff);
    }
    v16bf a0 = *(const v16bf*)&As[cur][wm * 32 + l][half * 16];
    v16bf a1 = *(const v16bf*)&As[cur][wm * 32 + 16 + l][half * 16];
    v16bf bf = *(const v16bf*)&Bs[cur][wn * 16 + l][half * 16];
    acc0 = __builtin_amdgcn_wmma_f32_16x16x32_bf16(false, a0, false, bf, (short)0, acc0, false, false);
    acc1 = __builtin_amdgcn_wmma_f32_16x16x32_bf16(false, a1, false, bf, (short)0, acc1, false, false);
    if (more) {
      #pragma unroll
      for (int j = 0; j < 4; ++j) *(v4bf*)&As[nxt][tmv[j]][tkv[j]] = ra[j];
      *(v4bf*)&Bs[nxt][bsn][bsk] = rb;
    }
  }

  const int   n_g = wn * 16 + l;
  const float bv  = bias[n_g];
  v8f accs[2] = {acc0, acc1};
  #pragma unroll
  for (int s = 0; s < 2; ++s)
    #pragma unroll
    for (int i = 0; i < 8; ++i) {
      int m = m_blk + wm * 32 + s * 16 + half * 8 + i;
      Y[(size_t)m * 32 + n_g] = accs[s][i] + bv;
    }
}

// ---------- NHWC conv (conv2/conv3): bf16 spikes in, fp32 pre-act out ----------
template<int CIN,int KH,int KW,int ST,int HIN,int WIN,int OH,int OW,int COUT>
__global__ __launch_bounds__(256) void conv_nhwc_wmma(
    const bf16_t* __restrict__ X, const bf16_t* __restrict__ W,
    const float* __restrict__ bias, float* __restrict__ Y)
{
  constexpr int P      = KH * KW;
  constexpr int K      = P * CIN;
  constexpr int KSTEPS = K / 32;
  constexpr int OHW    = OH * OW;
  __shared__ alignas(32) bf16_t As[2][128][48];
  __shared__ alignas(32) bf16_t Bs[2][64][48];

  const int tid  = threadIdx.x;
  const int lane = tid & 31, wave = tid >> 5;
  const int l    = lane & 15, half = lane >> 4;
  const int wm   = wave & 3,  wn   = wave >> 2;
  const int m_blk = blockIdx.x * 128;

  int tmv[2], tkv[2]; unsigned aoff[2];
  #pragma unroll
  for (int j = 0; j < 2; ++j) {
    int idx = tid + j * 256;
    int tm  = idx >> 2;
    int tk0 = (idx & 3) * 8;
    int m   = m_blk + tm;
    int nb  = m / OHW, rp = m - nb * OHW;
    int oh  = rp / OW, ow = rp - oh * OW;
    tmv[j] = tm; tkv[j] = tk0;
    aoff[j] = (unsigned)(((nb * HIN + oh * ST) * WIN + ow * ST) * CIN + tk0);
  }
  const unsigned woff = (unsigned)((tid >> 2) * K + (tid & 3) * 8);
  const int bsn = tid >> 2, bsk = (tid & 3) * 8;

  v8f zero = {};
  v8f acc[2][2] = {{zero, zero}, {zero, zero}};

  auto soff_of = [](int ks) -> int {
    int kb   = ks * 32;
    int pix  = kb / CIN;
    int coff = kb & (CIN - 1);
    int kh   = pix / KW, kw = pix - kh * KW;
    return (kh * WIN + kw) * CIN + coff;
  };

  // prologue: stage buffer 0
  {
    v8bf r0 = *(const v8bf*)(X + aoff[0]);
    v8bf r1 = *(const v8bf*)(X + aoff[1]);
    v8bf rb = *(const v8bf*)(W + woff);
    *(v8bf*)&As[0][tmv[0]][tkv[0]] = r0;
    *(v8bf*)&As[0][tmv[1]][tkv[1]] = r1;
    *(v8bf*)&Bs[0][bsn][bsk] = rb;
  }

  for (int ks = 0; ks < KSTEPS; ++ks) {
    const int cur = ks & 1, nxt = cur ^ 1;
    __syncthreads();
    v8bf r0 = {}, r1 = {}, rb = {};
    const bool more = (ks + 1) < KSTEPS;
    if (more) {
      const bf16_t* Xp = X + soff_of(ks + 1);        // uniform scalar base
      const bf16_t* Wp = W + (ks + 1) * 32;
      r0 = *(const v8bf*)(Xp + aoff[0]);
      r1 = *(const v8bf*)(Xp + aoff[1]);
      rb = *(const v8bf*)(Wp + woff);
    }
    v16bf a0 = *(const v16bf*)&As[cur][wm * 32 + l][half * 16];
    v16bf a1 = *(const v16bf*)&As[cur][wm * 32 + 16 + l][half * 16];
    #pragma unroll
    for (int t = 0; t < 2; ++t) {
      v16bf bfr = *(const v16bf*)&Bs[cur][(wn * 2 + t) * 16 + l][half * 16];
      acc[0][t] = __builtin_amdgcn_wmma_f32_16x16x32_bf16(false, a0, false, bfr, (short)0, acc[0][t], false, false);
      acc[1][t] = __builtin_amdgcn_wmma_f32_16x16x32_bf16(false, a1, false, bfr, (short)0, acc[1][t], false, false);
    }
    if (more) {
      *(v8bf*)&As[nxt][tmv[0]][tkv[0]] = r0;
      *(v8bf*)&As[nxt][tmv[1]][tkv[1]] = r1;
      *(v8bf*)&Bs[nxt][bsn][bsk] = rb;
    }
  }

  #pragma unroll
  for (int t = 0; t < 2; ++t) {
    int   n_g = (wn * 2 + t) * 16 + l;
    float bv  = bias[n_g];
    #pragma unroll
    for (int s = 0; s < 2; ++s)
      #pragma unroll
      for (int i = 0; i < 8; ++i) {
        int m = m_blk + wm * 32 + s * 16 + half * 8 + i;
        Y[(size_t)m * COUT + n_g] = acc[s][t][i] + bv;
      }
  }
}

// ---------- FC1 GEMM: Y[M,N] = A[M,K](bf16) * W[N,K]^T + bias, B tile via TDM ----------
__global__ __launch_bounds__(256) void gemm_wmma(
    const bf16_t* __restrict__ A, const bf16_t* __restrict__ W,
    const float* __restrict__ bias, float* __restrict__ Y,
    int M, int N, int K)
{
  (void)M;
  __shared__ alignas(32) bf16_t As[2][128][48];
  __shared__ alignas(32) bf16_t Bs[2][64][48];

  const int tid  = threadIdx.x;
  const int lane = tid & 31, wave = tid >> 5;
  const int l    = lane & 15, half = lane >> 4;
  const int wm   = wave & 3,  wn   = wave >> 2;
  const int m_blk = blockIdx.x * 128;
  const int n_blk = blockIdx.y * 64;

  const unsigned aoff0 = (unsigned)((m_blk + (tid >> 2)) * K + (tid & 3) * 8);
  const unsigned aoff1 = (unsigned)((m_blk + 64 + (tid >> 2)) * K + (tid & 3) * 8);
#if !HAVE_TDM
  const unsigned woff  = (unsigned)((n_blk + (tid >> 2)) * K + (tid & 3) * 8);
#endif
  const int bsn = tid >> 2, bsk = (tid & 3) * 8;
  const int KSTEPS = K / 32;

  v8f zero = {};
  v8f acc[2][2] = {{zero, zero}, {zero, zero}};

  // prologue: stage buffer 0
  {
#if HAVE_TDM
    if (wave == 0) tdm_load_b_tile(&Bs[0][0][0], W + (size_t)n_blk * K, K, N);
#else
    *(v8bf*)&Bs[0][bsn][bsk] = *(const v8bf*)(W + woff);
#endif
    v8bf r0 = *(const v8bf*)(A + aoff0);
    v8bf r1 = *(const v8bf*)(A + aoff1);
    *(v8bf*)&As[0][bsn][bsk] = r0;
    *(v8bf*)&As[0][64 + bsn][bsk] = r1;
#if HAVE_TDM
    if (wave == 0) __builtin_amdgcn_s_wait_tensorcnt(0);
#endif
  }

  for (int ks = 0; ks < KSTEPS; ++ks) {
    const int cur = ks & 1, nxt = cur ^ 1;
    __syncthreads();
    v8bf r0 = {}, r1 = {};
#if !HAVE_TDM
    v8bf rb = {};
#endif
    const bool more = (ks + 1) < KSTEPS;
    if (more) {
      const int kb = (ks + 1) * 32;
      const bf16_t* Ap = A + kb;
#if HAVE_TDM
      if (wave == 0) tdm_load_b_tile(&Bs[nxt][0][0], W + (size_t)n_blk * K + kb, K, N);
#else
      rb = *(const v8bf*)(W + woff + kb);
#endif
      r0 = *(const v8bf*)(Ap + aoff0);
      r1 = *(const v8bf*)(Ap + aoff1);
      __builtin_prefetch(A + aoff0 + kb + 32, 0, 0);
    }
    v16bf a0 = *(const v16bf*)&As[cur][wm * 32 + l][half * 16];
    v16bf a1 = *(const v16bf*)&As[cur][wm * 32 + 16 + l][half * 16];
    #pragma unroll
    for (int t = 0; t < 2; ++t) {
      v16bf bfr = *(const v16bf*)&Bs[cur][(wn * 2 + t) * 16 + l][half * 16];
      acc[0][t] = __builtin_amdgcn_wmma_f32_16x16x32_bf16(false, a0, false, bfr, (short)0, acc[0][t], false, false);
      acc[1][t] = __builtin_amdgcn_wmma_f32_16x16x32_bf16(false, a1, false, bfr, (short)0, acc[1][t], false, false);
    }
    if (more) {
      *(v8bf*)&As[nxt][bsn][bsk] = r0;
      *(v8bf*)&As[nxt][64 + bsn][bsk] = r1;
#if !HAVE_TDM
      *(v8bf*)&Bs[nxt][bsn][bsk] = rb;
#endif
#if HAVE_TDM
      if (wave == 0) __builtin_amdgcn_s_wait_tensorcnt(0);
#endif
    }
  }

  #pragma unroll
  for (int t = 0; t < 2; ++t) {
    int   n_g = n_blk + (wn * 2 + t) * 16 + l;
    float bv  = bias[n_g];
    #pragma unroll
    for (int s = 0; s < 2; ++s)
      #pragma unroll
      for (int i = 0; i < 8; ++i) {
        int m = m_blk + wm * 32 + s * 16 + half * 8 + i;
        Y[(size_t)m * N + n_g] = acc[s][t][i] + bv;
      }
  }
}

// ---------- LIF kernels (hard reset, tau=2, v_th=1, v_reset=0), x4 vectorized ----------
__global__ __launch_bounds__(256) void lif_bcast(const float* __restrict__ H,
                                                 bf16_t* __restrict__ S, int n) {
  int i = (blockIdx.x * 256 + threadIdx.x) * 4;
  if (i >= n) return;
  v4f x = *(const v4f*)(H + i);
  v4f v = {};
  #pragma unroll
  for (int t = 0; t < 8; ++t) {
    v4bf s4;
    #pragma unroll
    for (int e = 0; e < 4; ++e) {
      v[e] = v[e] + (x[e] - v[e]) * 0.5f;
      s4[e] = spike_bf(v[e]);
      v[e] = (v[e] >= 1.0f) ? 0.0f : v[e];
    }
    *(v4bf*)(S + (size_t)t * n + i) = s4;
  }
}

__global__ __launch_bounds__(256) void lif_seq(const float* __restrict__ Yp,
                                               bf16_t* __restrict__ S, int n) {
  int i = (blockIdx.x * 256 + threadIdx.x) * 4;
  if (i >= n) return;
  v4f v = {};
  #pragma unroll
  for (int t = 0; t < 8; ++t) {
    v4f x = *(const v4f*)(Yp + (size_t)t * n + i);
    v4bf s4;
    #pragma unroll
    for (int e = 0; e < 4; ++e) {
      v[e] = v[e] + (x[e] - v[e]) * 0.5f;
      s4[e] = spike_bf(v[e]);
      v[e] = (v[e] >= 1.0f) ? 0.0f : v[e];
    }
    *(v4bf*)(S + (size_t)t * n + i) = s4;
  }
}

__global__ __launch_bounds__(256) void lif_mean(const float* __restrict__ Yp,
                                                float* __restrict__ Mn, int n) {
  int i = (blockIdx.x * 256 + threadIdx.x) * 4;
  if (i >= n) return;
  v4f v = {}, acc = {};
  #pragma unroll
  for (int t = 0; t < 8; ++t) {
    v4f x = *(const v4f*)(Yp + (size_t)t * n + i);
    #pragma unroll
    for (int e = 0; e < 4; ++e) {
      v[e] = v[e] + (x[e] - v[e]) * 0.5f;
      acc[e] += (v[e] >= 1.0f) ? 1.0f : 0.0f;
      v[e] = (v[e] >= 1.0f) ? 0.0f : v[e];
    }
  }
  v4f outv;
  #pragma unroll
  for (int e = 0; e < 4; ++e) outv[e] = acc[e] * 0.125f;
  *(v4f*)(Mn + i) = outv;
}

// ---------- FC2 (tiny, fp32) ----------
__global__ __launch_bounds__(256) void fc2_kernel(const float* __restrict__ Mn,
                                                  const float* __restrict__ fw2,
                                                  const float* __restrict__ fb2,
                                                  float* __restrict__ out) {
  int idx = blockIdx.x * 256 + threadIdx.x;
  if (idx >= 512 * 6) return;
  int b = idx / 6, j = idx - b * 6;
  const float* mrow = Mn  + (size_t)b * 512;
  const float* wrow = fw2 + (size_t)j * 512;
  v4f a = {};
  #pragma unroll 4
  for (int k = 0; k < 128; ++k) {
    v4f mv = *(const v4f*)(mrow + k * 4);
    v4f wv = *(const v4f*)(wrow + k * 4);
    a += mv * wv;
  }
  out[idx] = fb2[j] + a[0] + a[1] + a[2] + a[3];
}

// ---------- launcher ----------
extern "C" void kernel_launch(void* const* d_in, const int* in_sizes, int n_in,
                              void* d_out, int out_size, void* d_ws, size_t ws_size,
                              hipStream_t stream) {
  (void)in_sizes; (void)n_in; (void)out_size; (void)ws_size;
  const float* x   = (const float*)d_in[0];
  const float* w1  = (const float*)d_in[1];
  const float* b1  = (const float*)d_in[2];
  const float* w2  = (const float*)d_in[3];
  const float* b2  = (const float*)d_in[4];
  const float* w3  = (const float*)d_in[5];
  const float* b3  = (const float*)d_in[6];
  const float* fw1 = (const float*)d_in[7];
  const float* fb1 = (const float*)d_in[8];
  const float* fw2 = (const float*)d_in[9];
  const float* fb2 = (const float*)d_in[10];

  size_t off = 0;
  auto take = [&](size_t bytes) -> char* {
    char* p = (char*)d_ws + off;
    off += (bytes + 255) & ~(size_t)255;
    return p;
  };
  bf16_t* W1b  = (bf16_t*)take((size_t)8192 * 2);
  bf16_t* W2b  = (bf16_t*)take((size_t)32768 * 2);
  bf16_t* W3b  = (bf16_t*)take((size_t)36864 * 2);
  bf16_t* FW1b = (bf16_t*)take((size_t)1605632 * 2);
  bf16_t* XB   = (bf16_t*)take((size_t)14450688 * 2);   // x as bf16 NCHW
  float*  H1   = (float*) take((size_t)6553600 * 4);    // [512,20,20,32] NHWC
  bf16_t* S1   = (bf16_t*)take((size_t)52428800 * 2);   // [8,512,20,20,32]
  float*  YY   = (float*) take((size_t)21233664 * 4);   // reusable pre-act
  bf16_t* S2   = (bf16_t*)take((size_t)21233664 * 2);   // [8,512,9,9,64]
  bf16_t* S3   = (bf16_t*)take((size_t)12845056 * 2);   // [8,512,7,7,64]
  float*  MEAN = (float*) take((size_t)262144 * 4);     // [512,512]

  cvt4_bf16  <<<8,     256, 0, stream>>>(w1, W1b, 8192);
  cvt4_bf16  <<<14112, 256, 0, stream>>>(x,  XB,  14450688);
  permute_cvt<<<128,   256, 0, stream>>>(w2,  W2b,  64,  32, 16);
  permute_cvt<<<144,   256, 0, stream>>>(w3,  W3b,  64,  64, 9);
  permute_cvt<<<6272,  256, 0, stream>>>(fw1, FW1b, 512, 64, 49);

  conv1_wmma<<<1600, 256, 0, stream>>>(XB, W1b, b1, H1);
  lif_bcast <<<6400, 256, 0, stream>>>(H1, S1, 6553600);

  conv_nhwc_wmma<32, 4, 4, 2, 20, 20, 9, 9, 64>
      <<<2592, 256, 0, stream>>>(S1, W2b, b2, YY);
  lif_seq<<<2592, 256, 0, stream>>>(YY, S2, 2654208);

  conv_nhwc_wmma<64, 3, 3, 1, 9, 9, 7, 7, 64>
      <<<1568, 256, 0, stream>>>(S2, W3b, b3, YY);
  lif_seq<<<1568, 256, 0, stream>>>(YY, S3, 1605632);

  gemm_wmma<<<dim3(32, 8), 256, 0, stream>>>(S3, FW1b, fb1, YY, 4096, 512, 3136);
  lif_mean<<<256, 256, 0, stream>>>(YY, MEAN, 262144);

  fc2_kernel<<<12, 256, 0, stream>>>(MEAN, fw2, fb2, (float*)d_out);
}
